// HypergraphNeuralNetwork_89421219103602
// MI455X (gfx1250) — matrix operand order repaired and checked by reading, not compiled, not run on hardware
//
#include <hip/hip_runtime.h>

// ---------------------------------------------------------------------------
// Hypergraph NN forward for MI455X (gfx1250, wave32, WMMA).
//   h   = LN(relu( Dinv * S_n( Binv * S_e( (x@W1) ) ) + b1 ))
//   out = Dinv * S_n( Binv * S_e( (h@W2) ) ) + b2 + x
// GEMMs use v_wmma_f32_16x16x32_bf16; A is converted f32->bf16 in-register
// with native bf16 converts, W is pre-transposed to bf16 once. Each wave owns
// a 16x128 output strip (8 accumulators) so one A fragment feeds 8 WMMAs.
// Sparse segment sums use f32 global atomics (edge accumulator = 16MB,
// resident in the 192MB L2).
// ---------------------------------------------------------------------------

#define N_NODES 20000
#define N_EDGES 4000
#define NNZ_CNT 200000
#define DIM_IN  512
#define DIM_HID 1024
#define DIM_OUT 512
#define LN_EPS  1e-5f

typedef __attribute__((ext_vector_type(4)))  float  v4f;
typedef __attribute__((ext_vector_type(8)))  float  v8f;
typedef __attribute__((ext_vector_type(16))) __bf16 v16bf;

__device__ __forceinline__ unsigned short f32_to_bf16_rne(float f) {
  unsigned int u = __float_as_uint(f);
  u += 0x7fffu + ((u >> 16) & 1u);           // round to nearest even
  return (unsigned short)(u >> 16);
}

// ------------------------------- utility kernels ---------------------------

__global__ void zero_f32_kernel(float* __restrict__ p, size_t n) {
  for (size_t i = (size_t)blockIdx.x * blockDim.x + threadIdx.x; i < n;
       i += (size_t)gridDim.x * blockDim.x)
    p[i] = 0.0f;
}

__global__ void degree_kernel(const int* __restrict__ ni, const int* __restrict__ hi,
                              const float* __restrict__ ew,
                              float* __restrict__ Dn, float* __restrict__ Be, int nnz) {
  int i = blockIdx.x * blockDim.x + threadIdx.x;
  if (i < nnz) {
    atomicAdd(&Dn[ni[i]], ew[hi[i]]);
    atomicAdd(&Be[hi[i]], 1.0f);
  }
}

__global__ void invert_kernel(float* __restrict__ p, int n) {
  int i = blockIdx.x * blockDim.x + threadIdx.x;
  if (i < n) { float v = p[i]; p[i] = (v > 0.0f) ? (1.0f / v) : 0.0f; }
}

// W [K x Nn] f32 row-major  ->  Wt [Nn x K] bf16 row-major (B column-major)
__global__ void transpose_cvt_bf16_kernel(const float* __restrict__ W,
                                          unsigned short* __restrict__ Wt,
                                          int K, int Nn) {
  size_t idx = (size_t)blockIdx.x * blockDim.x + threadIdx.x;
  size_t total = (size_t)K * Nn;
  if (idx >= total) return;
  int n = (int)(idx / K);
  int k = (int)(idx - (size_t)n * K);
  Wt[(size_t)n * K + k] = f32_to_bf16_rne(W[(size_t)k * Nn + n]);
}

// ------------------------------- WMMA GEMM ---------------------------------
// C[M,Nn] = A[M,K](f32, cvt->bf16 on the fly) * B, with Bt = B^T [Nn,K] bf16.
// 256 threads = 8 waves; each wave -> 16 rows x 128 cols (8 wmma accumulators;
// one A fragment reused across all 8 N-tiles each K-step).
__global__ __launch_bounds__(256) void gemm_bf16_wmma_kernel(
    const float* __restrict__ A, const unsigned short* __restrict__ Bt,
    float* __restrict__ C, int M, int K, int Nn) {
  const int lane    = threadIdx.x & 31;
  const int wave    = threadIdx.x >> 5;
  const int mBase   = blockIdx.x * 128 + wave * 16;
  const int nBase   = blockIdx.y * 128;
  if (mBase >= M) return;                       // wave-uniform: EXEC stays all-1s

  const int halfSel = lane >> 4;                // 0: K 0-15, 1: K 16-31
  const int rowA    = min(mBase + (lane & 15), M - 1);
  const float* arow = A + (size_t)rowA * K + 16 * halfSel;
  const __bf16* Btb = (const __bf16*)Bt;

  v8f acc[8] = {};
  for (int k0 = 0; k0 < K; k0 += 32) {
    // A fragment: lane holds one contiguous 16-element K-run 16*(lane>>4)..+15 (matches B).
    v4f a0 = *(const v4f*)(arow + k0);
    v4f a1 = *(const v4f*)(arow + k0 + 4);
    v4f a2 = *(const v4f*)(arow + k0 + 8);
    v4f a3 = *(const v4f*)(arow + k0 + 12);
    if (k0 + 32 < K) __builtin_prefetch(arow + k0 + 32, 0, 0);  // global_prefetch_b8

    v16bf af;
#pragma unroll
    for (int i = 0; i < 4; ++i) {               // native bf16 converts (packed cvt)
      af[i]      = (__bf16)a0[i];
      af[i + 4]  = (__bf16)a1[i];
      af[i + 8]  = (__bf16)a2[i];
      af[i + 12] = (__bf16)a3[i];
    }

#pragma unroll
    for (int t = 0; t < 8; ++t) {
      const int col = nBase + t * 16 + (lane & 15);
      // B 32x16 bf16: lane holds column N=lane&15, K-run 16*(lane>>4)..+15 -> one 32B load
      v16bf b = *(const v16bf*)(Btb + (size_t)col * K + k0 + 16 * halfSel);
      acc[t] = __builtin_amdgcn_wmma_f32_16x16x32_bf16(
          false, af, false, b, (short)0, acc[t], false, false);
    }
  }

#pragma unroll
  for (int t = 0; t < 8; ++t) {
    const int col = nBase + t * 16 + (lane & 15);
#pragma unroll
    for (int r = 0; r < 8; ++r) {               // C/D: VGPR r -> M=r+8*halfSel, N=lane&15
      const int row = mBase + r + 8 * halfSel;
      if (row < M) C[(size_t)row * Nn + col] = acc[t][r];
    }
  }
}

// ------------------------------ sparse scatter -----------------------------
// dst[didx[i]*D + f] += src[sidx[i]*D + f] * (sscale ? sscale[sidx[i]] : 1)
__global__ void scatter_add_kernel(const float* __restrict__ src,
                                   const int* __restrict__ sidx,
                                   const float* __restrict__ sscale,
                                   float* __restrict__ dst,
                                   const int* __restrict__ didx,
                                   int nnz, int dshift) {
  const long long D = 1ll << dshift;
  const long long total = (long long)nnz << dshift;
  for (long long t = (long long)blockIdx.x * blockDim.x + threadIdx.x; t < total;
       t += (long long)gridDim.x * blockDim.x) {
    int nz = (int)(t >> dshift);
    int f  = (int)(t & (D - 1));
    int s = sidx[nz], d = didx[nz];
    float v = src[((size_t)s << dshift) + f];
    if (sscale) v *= sscale[s];
    atomicAdd(&dst[((size_t)d << dshift) + f], v);
  }
}

// --------------------- fused scale+bias+relu+layernorm ---------------------
// In place on h [N_NODES, DIM_HID]: h = LN(relu(h*Dinv[row] + b1)) * g + b
__global__ __launch_bounds__(256) void node_post_ln_kernel(
    float* __restrict__ h, const float* __restrict__ Dinv,
    const float* __restrict__ b1, const float* __restrict__ g,
    const float* __restrict__ bln) {
  __shared__ float red[256];
  const int row = blockIdx.x;
  const int tid = threadIdx.x;
  float* hr = h + (size_t)row * DIM_HID;
  const float dinv = Dinv[row];

  float v[4], s = 0.f, ss = 0.f;
#pragma unroll
  for (int j = 0; j < 4; ++j) {
    int f = tid + j * 256;
    float t = fmaxf(hr[f] * dinv + b1[f], 0.0f);
    v[j] = t; s += t; ss += t * t;
  }
  red[tid] = s; __syncthreads();
  for (int k = 128; k > 0; k >>= 1) { if (tid < k) red[tid] += red[tid + k]; __syncthreads(); }
  float mean = red[0] * (1.0f / DIM_HID);
  __syncthreads();
  red[tid] = ss; __syncthreads();
  for (int k = 128; k > 0; k >>= 1) { if (tid < k) red[tid] += red[tid + k]; __syncthreads(); }
  float var  = red[0] * (1.0f / DIM_HID) - mean * mean;
  float rstd = rsqrtf(var + LN_EPS);
#pragma unroll
  for (int j = 0; j < 4; ++j) {
    int f = tid + j * 256;
    hr[f] = (v[j] - mean) * rstd * g[f] + bln[f];
  }
}

// out = agg2*Dinv[row] + b2[col] + x  (residual fused)
__global__ void final_residual_kernel(const float* __restrict__ agg2,
                                      const float* __restrict__ Dinv,
                                      const float* __restrict__ b2,
                                      const float* __restrict__ x,
                                      float* __restrict__ out) {
  const size_t total = (size_t)N_NODES * DIM_OUT;
  for (size_t t = (size_t)blockIdx.x * blockDim.x + threadIdx.x; t < total;
       t += (size_t)gridDim.x * blockDim.x) {
    int row = (int)(t >> 9);
    int col = (int)(t & 511);
    out[t] = agg2[t] * Dinv[row] + b2[col] + x[t];
  }
}

// ------------------------------- launcher ----------------------------------

extern "C" void kernel_launch(void* const* d_in, const int* in_sizes, int n_in,
                              void* d_out, int out_size, void* d_ws, size_t ws_size,
                              hipStream_t stream) {
  (void)in_sizes; (void)n_in; (void)out_size; (void)ws_size;

  const float* x   = (const float*)d_in[0];
  const int*   ni  = (const int*)d_in[1];
  const int*   hi  = (const int*)d_in[2];
  const float* ew  = (const float*)d_in[3];
  const float* W1  = (const float*)d_in[4];
  const float* b1  = (const float*)d_in[5];
  const float* W2  = (const float*)d_in[6];
  const float* b2  = (const float*)d_in[7];
  const float* lng = (const float*)d_in[8];
  const float* lnb = (const float*)d_in[9];
  float* out = (float*)d_out;

  // Workspace layout (256B aligned regions, reused across stages).
  char* p = (char*)d_ws;
  auto alloc = [&](size_t bytes) -> char* {
    char* r = p; p += (bytes + 255) & ~(size_t)255; return r;
  };
  float*          Dn    = (float*)alloc((size_t)(N_NODES + N_EDGES) * 4);
  float*          Be    = Dn + N_NODES;                       // contiguous with Dn
  unsigned short* W1t   = (unsigned short*)alloc((size_t)DIM_HID * DIM_IN * 2);
  unsigned short* W2t   = (unsigned short*)alloc((size_t)DIM_OUT * DIM_HID * 2);
  float*          bufXT = (float*)alloc((size_t)N_NODES * DIM_HID * 4); // xt; later yt+agg2
  float*          bufEH = (float*)alloc((size_t)N_EDGES * DIM_HID * 4); // eh1; later eh2
  float*          bufAG = (float*)alloc((size_t)N_NODES * DIM_HID * 4); // agg1 -> h (in place)
  float* xt   = bufXT;
  float* eh1  = bufEH;
  float* agg1 = bufAG;
  float* yt   = bufXT;                                        // xt dead after scatter1
  float* agg2 = bufXT + (size_t)N_NODES * DIM_OUT;
  float* eh2  = bufEH;                                        // eh1 dead after scatter2

  // Degrees: Dn = sum w_e over incidences, Be = |incidences|; then invert in place.
  zero_f32_kernel<<<1024, 256, 0, stream>>>(Dn, (size_t)(N_NODES + N_EDGES));
  degree_kernel<<<(NNZ_CNT + 255) / 256, 256, 0, stream>>>(ni, hi, ew, Dn, Be, NNZ_CNT);
  invert_kernel<<<(N_NODES + N_EDGES + 255) / 256, 256, 0, stream>>>(Dn, N_NODES + N_EDGES);

  // Weight prep: transpose + bf16 (tiny, L2-resident, reused 1250x by the GEMMs).
  transpose_cvt_bf16_kernel<<<(unsigned)(((size_t)DIM_IN * DIM_HID + 255) / 256), 256, 0,
                              stream>>>(W1, W1t, DIM_IN, DIM_HID);
  transpose_cvt_bf16_kernel<<<(unsigned)(((size_t)DIM_HID * DIM_OUT + 255) / 256), 256, 0,
                              stream>>>(W2, W2t, DIM_HID, DIM_OUT);

  // ---- layer 1 ----
  dim3 g1((N_NODES + 127) / 128, DIM_HID / 128);
  gemm_bf16_wmma_kernel<<<g1, 256, 0, stream>>>(x, W1t, xt, N_NODES, DIM_IN, DIM_HID);
  zero_f32_kernel<<<4096, 256, 0, stream>>>(eh1, (size_t)N_EDGES * DIM_HID);
  scatter_add_kernel<<<4096, 256, 0, stream>>>(xt, ni, nullptr, eh1, hi, NNZ_CNT, 10);
  zero_f32_kernel<<<4096, 256, 0, stream>>>(agg1, (size_t)N_NODES * DIM_HID);
  scatter_add_kernel<<<4096, 256, 0, stream>>>(eh1, hi, Be, agg1, ni, NNZ_CNT, 10);
  node_post_ln_kernel<<<N_NODES, 256, 0, stream>>>(agg1, Dn, b1, lng, lnb);

  // ---- layer 2 ----
  dim3 g2((N_NODES + 127) / 128, DIM_OUT / 128);
  gemm_bf16_wmma_kernel<<<g2, 256, 0, stream>>>(agg1, W2t, yt, N_NODES, DIM_HID, DIM_OUT);
  zero_f32_kernel<<<4096, 256, 0, stream>>>(eh2, (size_t)N_EDGES * DIM_OUT);
  scatter_add_kernel<<<4096, 256, 0, stream>>>(yt, ni, nullptr, eh2, hi, NNZ_CNT, 9);
  zero_f32_kernel<<<4096, 256, 0, stream>>>(agg2, (size_t)N_NODES * DIM_OUT);
  scatter_add_kernel<<<4096, 256, 0, stream>>>(eh2, hi, Be, agg2, ni, NNZ_CNT, 9);
  final_residual_kernel<<<4096, 256, 0, stream>>>(agg2, Dn, b2, x, out);
}
